// HyperFNS_88304527606511
// MI455X (gfx1250) — compile-verified, matrix operand order checked by reading, not run on hardware
//
#include <hip/hip_runtime.h>
#include <hip/hip_bf16.h>

// ---------------------------------------------------------------------------
// HyperFNS on MI455X (gfx1250, wave32).
//  - Darcy stencil / Jacobi: LDS-tiled, bandwidth-bound (L2-resident).
//  - iFFT2(1024) / FFT2(1025): DFT-as-GEMM on the matrix pipe with
//    v_wmma_f32_16x16x32_bf16; twiddles via __sincosf (TRANS, co-executes).
//  - shift/crop/pad folded into index maps; complex 3x3 convs on VALU.
// ---------------------------------------------------------------------------

typedef __attribute__((ext_vector_type(16))) __bf16 v16bf;
typedef __attribute__((ext_vector_type(8)))  float  v8f;

#define Bn   4
#define Gg   513
#define G2   (513*513)
#define NCg  512
#define NC2  (512*512)
#define Ig   511
#define I2   (511*511)
#define F24  (1024*1024)
#define F25  (1025*1025)
#define PI2F 6.28318530717958647692f

// ---------------------------------------------------------------- utilities
__global__ void k_zero(float* __restrict__ p, int n) {
  int i = blockIdx.x * 256 + threadIdx.x;
  if (i < n) p[i] = 0.f;
}

// dinv = 1 / conv2x2(coef, 2/3)   : (B,511,511)
__global__ void k_dinv(const float* __restrict__ a, float* __restrict__ d) {
  int idx = blockIdx.x * 256 + threadIdx.x;
  if (idx >= Bn * I2) return;
  int b = idx / I2, t = idx % I2, y = t / Ig, x = t % Ig;
  const float* ab = a + (size_t)b * NC2;
  float s = ab[y*NCg + x] + ab[y*NCg + x + 1] + ab[(y+1)*NCg + x] + ab[(y+1)*NCg + x + 1];
  d[idx] = 1.f / ((2.f / 3.f) * s);
}

// 9-point variable-coefficient Darcy stencil at interior point.
// xs[r][c] holds x(gy0+r, gx0+c) with the convention that the output point
// (oy,ox) maps to xs[ty+1][tx+1]; (ay,ax) = (oy-1, ox-1) top-left 'a' index.
__device__ __forceinline__ float darcy_pt(const float xs[18][19],
                                          const float* __restrict__ ab,
                                          int ty, int tx, int ay, int ax) {
  const float K0[2][2] = {{-1.f/6.f,  2.f/3.f}, {-1.f/3.f, -1.f/6.f}};
  const float K1[2][2] = {{ 2.f/3.f, -1.f/6.f}, {-1.f/6.f, -1.f/3.f}};
  const float K2[2][2] = {{-1.f/6.f, -1.f/3.f}, { 2.f/3.f, -1.f/6.f}};
  const float K3[2][2] = {{-1.f/3.f, -1.f/6.f}, {-1.f/6.f,  2.f/3.f}};
  float acc = 0.f;
#pragma unroll
  for (int p = 0; p < 2; ++p)
#pragma unroll
    for (int q = 0; q < 2; ++q) {
      float av = ab[(ay + p) * NCg + ax + q];
      acc += av * (K0[p][q] * xs[ty+p+1][tx+q]     +
                   K1[p][q] * xs[ty+p+1][tx+q+1]   +
                   K2[p][q] * xs[ty+p  ][tx+q+1]   +
                   K3[p][q] * xs[ty+p  ][tx+q]);
    }
  return acc;
}

// one weighted-Jacobi sweep (interior only; borders stay zero)
__global__ void k_jacobi(const float* __restrict__ xin, const float* __restrict__ a,
                         const float* __restrict__ f,  const float* __restrict__ dinv,
                         float* __restrict__ xout) {
  __shared__ float xs[18][19];
  int b = blockIdx.z, ty = threadIdx.y, tx = threadIdx.x;
  int gy0 = blockIdx.y * 16, gx0 = blockIdx.x * 16;
  const float* xb = xin + (size_t)b * G2;
  int tid = ty * 16 + tx;
  for (int t = tid; t < 18 * 18; t += 256) {
    int r = t / 18, c = t % 18;
    int yy = gy0 + r, xx = gx0 + c;
    xs[r][c] = (yy < Gg && xx < Gg) ? xb[yy*Gg + xx] : 0.f;
  }
  __syncthreads();
  int oy = gy0 + ty + 1, ox = gx0 + tx + 1;
  if (oy > Ig || ox > Ig) return;
  float Ax = darcy_pt(xs, a + (size_t)b * NC2, ty, tx, oy - 1, ox - 1);
  float rr = f[(size_t)b*G2 + oy*Gg + ox] - Ax;
  xout[(size_t)b*G2 + oy*Gg + ox] =
      xs[ty+1][tx+1] + 0.75f * dinv[(size_t)b*I2 + (oy-1)*Ig + (ox-1)] * rr;
}

// out = (f ? f - A x : A x), full 513^2 grid (border: Ax = 0)
__global__ void k_darcy(const float* __restrict__ x, const float* __restrict__ a,
                        const float* __restrict__ f, float* __restrict__ out) {
  __shared__ float xs[18][19];
  int b = blockIdx.z, ty = threadIdx.y, tx = threadIdx.x;
  int oy0 = blockIdx.y * 16, ox0 = blockIdx.x * 16;
  const float* xb = x + (size_t)b * G2;
  int tid = ty * 16 + tx;
  for (int t = tid; t < 18 * 18; t += 256) {
    int r = t / 18, c = t % 18;
    int yy = oy0 - 1 + r, xx = ox0 - 1 + c;
    xs[r][c] = (yy >= 0 && yy < Gg && xx >= 0 && xx < Gg) ? xb[yy*Gg + xx] : 0.f;
  }
  __syncthreads();
  int oy = oy0 + ty, ox = ox0 + tx;
  if (oy >= Gg || ox >= Gg) return;
  float Ax = 0.f;
  if (oy >= 1 && oy <= Ig && ox >= 1 && ox <= Ig)
    Ax = darcy_pt(xs, a + (size_t)b * NC2, ty, tx, oy - 1, ox - 1);
  size_t o = (size_t)b*G2 + (size_t)oy*Gg + ox;
  out[o] = f ? (f[o] - Ax) : Ax;
}

// symmetric odd expansion of r[1:-1,1:-1] (511^2) to 1024^2 (stride 1024, slab F25)
__global__ void k_expand(const float* __restrict__ r, float* __restrict__ zre,
                         float* __restrict__ zim) {
  int idx = blockIdx.x * 256 + threadIdx.x;
  if (idx >= Bn * F24) return;
  int b = idx / F24, t = idx % F24, u = t >> 10, v = t & 1023;
  float sy = 1.f, sx = 1.f; int iy = -1, ix = -1;
  if (u >= 1 && u <= 511)      iy = u - 1;
  else if (u >= 513)         { iy = 1023 - u; sy = -1.f; }
  if (v >= 1 && v <= 511)      ix = v - 1;
  else if (v >= 513)         { ix = 1023 - v; sx = -1.f; }
  float val = 0.f;
  if (iy >= 0 && ix >= 0)
    val = sy * sx * r[(size_t)b*G2 + (size_t)(iy+1)*Gg + (ix+1)];
  zre[(size_t)b*F25 + t] = val;
  zim[(size_t)b*F25 + t] = 0.f;
}

// --------------------------------------------------------- DFT-as-GEMM (WMMA)
// out[c][m] = scale * sum_n in[m][n] * exp(i*ang_unit*cc*n),  cc=(c+c_add)%c_mod
// One 16x16 complex tile per wave; K-chunks of 32 (bf16 A/B, f32 acc).
__launch_bounds__(256)
__global__ void k_cdft(const float* __restrict__ in_re, const float* __restrict__ in_im,
                       float* __restrict__ out_re, float* __restrict__ out_im,
                       int M, int N, int Cout, int in_rstride, int in_bstride,
                       int out_stride, int out_bstride,
                       float ang_unit, float scale, int c_add, int c_mod) {
  __shared__ float lsR[16][33];
  __shared__ float lsI[16][33];
  int tid  = threadIdx.x;
  int lane = tid & 31, wave = tid >> 5;
  int b  = blockIdx.z;
  int m0 = blockIdx.y * 16;
  int c0 = blockIdx.x * 128 + wave * 16;
  int colN = lane & 15;
  int hi   = (lane >> 4) & 1;
  int cg   = (c0 + colN + c_add) % c_mod;
  float ang = ang_unit * (float)cg;
  v8f acc_re = {}; v8f acc_im = {};
  const float* ibr = in_re + (size_t)b * in_bstride;
  const float* ibi = in_im + (size_t)b * in_bstride;
  int nch = (N + 31) >> 5;
  for (int kc = 0; kc < nch; ++kc) {
    int nb = kc << 5;
    for (int t = tid; t < 1024; t += 256) {             // stage 16x32 re + im
      int pl = t >> 9, rr = (t >> 5) & 15, cc = t & 31;
      int n = nb + cc, row = m0 + rr;
      float v = 0.f;
      if (n < N && row < M) {
        const float* src = pl ? ibi : ibr;
        v = src[(size_t)row * in_rstride + n];
        if (n + 32 < N) __builtin_prefetch(&src[(size_t)row * in_rstride + n + 32], 0, 0);
      }
      (pl ? lsI : lsR)[rr][cc] = v;
    }
    __syncthreads();
    v16bf are, aim, ain, bre, bim;
    int arow = lane & 15;
#pragma unroll
    for (int e = 0; e < 16; ++e) {
      int k = hi * 8 + ((e >> 3) << 4) + (e & 7);       // A/B 16-bit K striping
      float vr = lsR[arow][k], vi = lsI[arow][k];
      are[e] = (__bf16)vr;
      aim[e] = (__bf16)vi;
      ain[e] = (__bf16)(-vi);                           // bf16 WMMA forbids A-neg
      float s, c;
      __sincosf(ang * (float)(nb + k), &s, &c);
      bre[e] = (__bf16)c;
      bim[e] = (__bf16)s;
    }
    acc_re = __builtin_amdgcn_wmma_f32_16x16x32_bf16(false, are, false, bre, (short)0, acc_re, false, false);
    acc_re = __builtin_amdgcn_wmma_f32_16x16x32_bf16(false, ain, false, bim, (short)0, acc_re, false, false);
    acc_im = __builtin_amdgcn_wmma_f32_16x16x32_bf16(false, are, false, bim, (short)0, acc_im, false, false);
    acc_im = __builtin_amdgcn_wmma_f32_16x16x32_bf16(false, aim, false, bre, (short)0, acc_im, false, false);
    __syncthreads();
  }
  int col = c0 + colN;                                  // transposed store
  if (col < Cout) {
#pragma unroll
    for (int j = 0; j < 8; ++j) {
      int row = m0 + j + hi * 8;
      if (row < M) {
        size_t o = (size_t)b * out_bstride + (size_t)col * out_stride + row;
        out_re[o] = acc_re[j] * scale;
        out_im[o] = acc_im[j] * scale;
      }
    }
  }
}

// per-sample complex 3x3 conv, pad=1; transposed=1 => conjT(w) semantics
__global__ void k_mcc(const float* __restrict__ xr, const float* __restrict__ xi,
                      float* __restrict__ yr, float* __restrict__ yi,
                      const float* __restrict__ wr, const float* __restrict__ wi,
                      int Cin, int Cout, int transposed) {
  __shared__ float lwr[4*4*9], lwi[4*4*9];
  int b = blockIdx.z;
  int tid = threadIdx.y * 16 + threadIdx.x;
  int nw = Cout * Cin * 9;
  for (int t = tid; t < nw; t += 256) {
    int co = t / (Cin * 9), rem = t % (Cin * 9), ci = rem / 9, tap = rem % 9;
    int ky = tap / 3, kx = tap % 3;
    size_t si; float sgn = 1.f;
    if (!transposed) si = (((size_t)b * Cout + co) * Cin + ci) * 9 + ky * 3 + kx;
    else { si = (((size_t)b * Cin + ci) * Cout + co) * 9 + kx * 3 + ky; sgn = -1.f; }
    lwr[t] = wr[si];
    lwi[t] = sgn * wi[si];
  }
  __syncthreads();
  int ox = blockIdx.x * 16 + threadIdx.x, oy = blockIdx.y * 16 + threadIdx.y;
  if (ox >= Gg || oy >= Gg) return;
  for (int co = 0; co < Cout; ++co) {
    float ar = 0.f, ai = 0.f;
    for (int ci = 0; ci < Cin; ++ci) {
      const float* pr = xr + ((size_t)b * Cin + ci) * G2;
      const float* pi = xi + ((size_t)b * Cin + ci) * G2;
#pragma unroll
      for (int ky = 0; ky < 3; ++ky) {
        int iy = oy + ky - 1; if (iy < 0 || iy >= Gg) continue;
#pragma unroll
        for (int kx = 0; kx < 3; ++kx) {
          int ix = ox + kx - 1; if (ix < 0 || ix >= Gg) continue;
          float vr = pr[iy*Gg + ix], vi = pi[iy*Gg + ix];
          float cr = lwr[(co*Cin + ci)*9 + ky*3 + kx];
          float cm = lwi[(co*Cin + ci)*9 + ky*3 + kx];
          ar += vr * cr - vi * cm;
          ai += vr * cm + vi * cr;
        }
      }
    }
    size_t o = ((size_t)b * Cout + co) * G2 + (size_t)oy * Gg + ox;
    yr[o] = ar; yi[o] = ai;
  }
}

// z *= (wtr + i*wti) * ik2  (channel-0 of a zstride-strided tensor)
__global__ void k_ptwise(float* __restrict__ zr, float* __restrict__ zi,
                         const float* __restrict__ wtr, const float* __restrict__ wti,
                         int zstride) {
  int idx = blockIdx.x * 256 + threadIdx.x;
  if (idx >= Bn * G2) return;
  int b = idx / G2, t = idx % G2, y = t / Gg, x = t % Gg;
  float kx = (float)(x - 256) * 3.14159265358979f;
  float ky = (float)(y - 256) * 3.14159265358979f;
  float ik2 = (x == 256 && y == 256) ? 1.f : 1.f / (kx*kx + ky*ky);
  float wr = wtr[idx] * ik2, wi = wti[idx] * ik2;
  size_t o = (size_t)b * zstride + t;
  float a = zr[o], c = zi[o];
  zr[o] = a * wr - c * wi;
  zi[o] = a * wi + c * wr;
}

// zero-pad 513^2 -> 1025^2 with ifftshift folded in: src idx = (u+255)%1025
__global__ void k_pack(const float* __restrict__ sr, const float* __restrict__ si,
                       float* __restrict__ dr, float* __restrict__ di, int in_bstride) {
  int idx = blockIdx.x * 256 + threadIdx.x;
  if (idx >= Bn * F25) return;
  int b = idx / F25, t = idx % F25, u = t / 1025, v = t % 1025;
  int iy = (u + 255) % 1025, ix = (v + 255) % 1025;
  float vr = 0.f, vi = 0.f;
  if (iy < Gg && ix < Gg) {
    size_t s = (size_t)b * in_bstride + (size_t)iy * Gg + ix;
    vr = sr[s]; vi = si[s];
  }
  dr[idx] = vr; di[idx] = vi;
}

// e (513^2, zero border) <- real crop of fft output (511^2)
__global__ void k_emit(const float* __restrict__ t2r, float* __restrict__ e) {
  int idx = blockIdx.x * 256 + threadIdx.x;
  if (idx >= Bn * G2) return;
  int b = idx / G2, t = idx % G2, y = t / Gg, x = t % Gg;
  float v = 0.f;
  if (y >= 1 && y <= Ig && x >= 1 && x <= Ig)
    v = t2r[(size_t)b * I2 + (size_t)(y-1) * Ig + (x-1)];
  e[idx] = v;
}

// per-slot dot product with wave32 shuffle + LDS + atomic reduction
__global__ void k_dot(const float* __restrict__ p, const float* __restrict__ q,
                      float* __restrict__ out, int n, int bstride) {
  int b = blockIdx.y;
  const float* pp = p + (size_t)b * bstride;
  const float* qq = q + (size_t)b * bstride;
  float s = 0.f;
  for (int i = blockIdx.x * 256 + threadIdx.x; i < n; i += gridDim.x * 256)
    s += pp[i] * qq[i];
#pragma unroll
  for (int o = 16; o; o >>= 1) s += __shfl_xor(s, o, 32);
  __shared__ float ws[8];
  int lane = threadIdx.x & 31, wv = threadIdx.x >> 5;
  if (!lane) ws[wv] = s;
  __syncthreads();
  if (threadIdx.x == 0) {
    float t = 0.f;
#pragma unroll
    for (int i = 0; i < 8; ++i) t += ws[i];
    atomicAdd(&out[b], t);
  }
}

__global__ void k_upd(float* __restrict__ x, const float* __restrict__ e,
                      const float* __restrict__ sums) {
  int idx = blockIdx.x * 256 + threadIdx.x;
  if (idx >= Bn * G2) return;
  int b = idx / G2;
  x[idx] += (sums[b] / sums[8 + b]) * e[idx];
}

__global__ void k_final(const float* __restrict__ sums, float* __restrict__ out) {
  if (threadIdx.x == 0 && blockIdx.x == 0) out[0] = sqrtf(sums[16] / sums[17]);
}

// ---------------------------------------------------------------------------
extern "C" void kernel_launch(void* const* d_in, const int* in_sizes, int n_in,
                              void* d_out, int out_size, void* d_ws, size_t ws_size,
                              hipStream_t stream) {
  (void)in_sizes; (void)n_in; (void)out_size; (void)ws_size;
  const float* f    = (const float*)d_in[0];
  const float* coef = (const float*)d_in[1];
  const float* w1r  = (const float*)d_in[3];
  const float* w1i  = (const float*)d_in[4];
  const float* w2r  = (const float*)d_in[5];
  const float* w2i  = (const float*)d_in[6];
  const float* w3r  = (const float*)d_in[7];
  const float* w3i  = (const float*)d_in[8];
  const float* wtr  = (const float*)d_in[9];
  const float* wti  = (const float*)d_in[10];
  const int K = 2;  // harness epoch = 101 -> K = (101-1)/100 + 1

  float* W = (float*)d_ws;
  size_t o = 0;
  float* x0   = W + o; o += (size_t)Bn * G2;
  float* x1   = W + o; o += (size_t)Bn * G2;
  float* r_   = W + o; o += (size_t)Bn * G2;
  float* e_   = W + o; o += (size_t)Bn * G2;
  float* Ae   = W + o; o += (size_t)Bn * G2;
  float* dinv = W + o; o += (size_t)Bn * I2;
  float* sums = W + o; o += 32;
  float* fr   = W + o; o += (size_t)Bn * F25;   // rsym / packed-1025 (phases disjoint)
  float* fi   = W + o; o += (size_t)Bn * F25;
  float* t1r  = W + o; o += (size_t)Bn * F25;
  float* t1i  = W + o; o += (size_t)Bn * F25;
  float* t2r  = W + o; o += (size_t)Bn * I2;
  float* t2i  = W + o; o += (size_t)Bn * I2;
  float* c0r  = W + o; o += (size_t)Bn * 4 * G2;
  float* c0i  = W + o; o += (size_t)Bn * 4 * G2;
  float* c1r  = W + o; o += (size_t)Bn * 4 * G2;
  float* c1i  = W + o; o += (size_t)Bn * 4 * G2;

  auto zero = [&](float* p, size_t n) {
    k_zero<<<(int)((n + 255) / 256), 256, 0, stream>>>(p, (int)n);
  };

  zero(x0, (size_t)Bn * G2);
  zero(x1, (size_t)Bn * G2);
  k_dinv<<<(Bn * I2 + 255) / 256, 256, 0, stream>>>(coef, dinv);

  dim3 tb(16, 16);
  dim3 gInt(32, 32, Bn);
  dim3 gFull(33, 33, Bn);

  const float angI = +PI2F / 1024.f;   // ifft
  const float angF = -PI2F / 1025.f;   // fft

  float* xc = x0;
  float* xn = x1;
  for (int step = 0; step < K; ++step) {
    for (int it = 0; it < 10; ++it) {
      k_jacobi<<<gInt, tb, 0, stream>>>(xc, coef, f, dinv, xn);
      float* tmp = xc; xc = xn; xn = tmp;
    }
    k_darcy<<<gFull, tb, 0, stream>>>(xc, coef, f, r_);      // r = f - A x

    // ---- H_apply: ifft2(1024) with fftshift+crop folded (cc=(c+768)%1024)
    k_expand<<<(Bn * F24 + 255) / 256, 256, 0, stream>>>(r_, fr, fi);
    k_cdft<<<dim3(5, 64, Bn), 256, 0, stream>>>(fr, fi, t1r, t1i,
        1024, 1024, 513, 1024, F25, 1024, F25, angI, 1.f / 1024.f, 768, 1024);
    k_cdft<<<dim3(5, 33, Bn), 256, 0, stream>>>(t1r, t1i, c0r, c0i,
        513, 1024, 513, 1024, F25, 513, 4 * G2, angI, 1.f / 1024.f, 768, 1024);

    // ---- complex conv chain + spectral weight
    k_mcc<<<gFull, tb, 0, stream>>>(c0r, c0i, c1r, c1i, w1r, w1i, 1, 4, 0);
    k_mcc<<<gFull, tb, 0, stream>>>(c1r, c1i, c0r, c0i, w2r, w2i, 4, 4, 0);
    k_mcc<<<gFull, tb, 0, stream>>>(c0r, c0i, c1r, c1i, w3r, w3i, 4, 1, 0);
    k_ptwise<<<(Bn * G2 + 255) / 256, 256, 0, stream>>>(c1r, c1i, wtr, wti, 4 * G2);
    k_mcc<<<gFull, tb, 0, stream>>>(c1r, c1i, c0r, c0i, w3r, w3i, 1, 4, 1);
    k_mcc<<<gFull, tb, 0, stream>>>(c0r, c0i, c1r, c1i, w2r, w2i, 4, 4, 1);
    k_mcc<<<gFull, tb, 0, stream>>>(c1r, c1i, c0r, c0i, w1r, w1i, 4, 1, 1);

    // ---- fft2(1025) with ifftshift folded into pack; crop to 511^2
    k_pack<<<(Bn * F25 + 255) / 256, 256, 0, stream>>>(c0r, c0i, fr, fi, 4 * G2);
    k_cdft<<<dim3(4, 65, Bn), 256, 0, stream>>>(fr, fi, t1r, t1i,
        1025, 1025, 511, 1025, F25, 1025, F25, angF, 1.f, 0, 1025);
    k_cdft<<<dim3(4, 32, Bn), 256, 0, stream>>>(t1r, t1i, t2r, t2i,
        511, 1025, 511, 1025, F25, 511, I2, angF, 1.f, 0, 1025);
    k_emit<<<(Bn * G2 + 255) / 256, 256, 0, stream>>>(t2r, e_);

    // ---- line search: alpha = <r,e> / <Ae,e>
    k_darcy<<<gFull, tb, 0, stream>>>(e_, coef, nullptr, Ae);
    zero(sums, 32);
    k_dot<<<dim3(64, Bn), 256, 0, stream>>>(r_, e_, sums + 0, G2, G2);
    k_dot<<<dim3(64, Bn), 256, 0, stream>>>(Ae, e_, sums + 8, G2, G2);
    k_upd<<<(Bn * G2 + 255) / 256, 256, 0, stream>>>(xc, e_, sums);
  }

  k_darcy<<<gFull, tb, 0, stream>>>(xc, coef, f, r_);
  zero(sums, 32);
  k_dot<<<dim3(256, 1), 256, 0, stream>>>(r_, r_, sums + 16, Bn * G2, 0);
  k_dot<<<dim3(256, 1), 256, 0, stream>>>(f,  f,  sums + 17, Bn * G2, 0);
  k_final<<<1, 32, 0, stream>>>(sums, (float*)d_out);
}
